// DecoderLayer_2121713845000
// MI455X (gfx1250) — matrix operand order repaired and checked by reading, not compiled, not run on hardware
//
#include <hip/hip_runtime.h>
#include <hip/hip_bf16.h>

// ---------------------------------------------------------------------------
// CDNA5 (gfx1250) transformer decoder layer, bf16 WMMA + fp32 accumulate.
// v2: double-buffered LDS pipeline (1 barrier/chunk), 32x64 wave tiles.
// ---------------------------------------------------------------------------

typedef __bf16 bf16_t;
typedef __attribute__((ext_vector_type(16))) __bf16 v16bf;
typedef __attribute__((ext_vector_type(8)))  float  v8f;

#define DMODEL 1024
#define DFF    4096
#define SEQ    2048
#define BATCH  4
#define NTOK   (BATCH * SEQ)

// GEMM tiling: block computes BM x BN, K-chunked by BK, double-buffered.
// 8 waves (256 thr) arranged 4x2; each wave owns 32x64 = 2x4 WMMA tiles.
#define BM 128
#define BN 128
#define BK 32
#define LDS_STRIDE (BK + 8)   // bf16 elems per LDS row (pad vs bank conflicts)

enum {
  MODE_BF16       = 0,  // C bf16 = acc                       (Q, K proj)
  MODE_BF16_TRANS = 1,  // Ct[n,m] bf16 = acc                 (V^T)
  MODE_SCORES     = 2,  // C f32 = causal(acc*scale)          (Q K^T)
  MODE_CTX        = 3,  // C bf16 = acc, K-loop causal-limit  (P V)
  MODE_RESID      = 4,  // C f32 = acc + resid                (W_o + x)
  MODE_RELU_BIAS  = 5,  // C bf16 = relu(acc + bias)          (fc1)
  MODE_BIAS_RESID = 6,  // C f32 = acc + bias + resid         (fc2 + x1)
};

// ---------------------------------------------------------------------------
// Tiled bf16 WMMA GEMM: out[m,n] = sum_k A[m,k] * Bt[n,k]   (+ epilogue)
// A: [M x K] row-major bf16, lda; Bt: [N x K] row-major bf16, ldb.
// M % BM == 0, N % BN == 0, K % BK == 0 (true for all shapes here).
// ---------------------------------------------------------------------------
template <int MODE>
__global__ __launch_bounds__(256)
void gemm_wmma(const bf16_t* __restrict__ A, int lda,
               const bf16_t* __restrict__ Bt, int ldb,
               void* __restrict__ Cout, int ldc,
               const float* __restrict__ bias,
               const float* __restrict__ resid,
               int M, int N, int K, float scale)
{
  __shared__ bf16_t As[2][BM * LDS_STRIDE];
  __shared__ bf16_t Bs[2][BN * LDS_STRIDE];

  const int tid     = threadIdx.x;
  const int lane    = tid & 31;
  const int wave    = tid >> 5;       // 0..7
  const int waveM   = wave >> 1;      // 0..3 -> 32-row strip
  const int waveN   = wave & 1;       // 0..1 -> 64-col strip
  const int halfSel = lane >> 4;      // 0 = lanes 0-15, 1 = lanes 16-31
  const int l16     = lane & 15;

  const int rowBase = blockIdx.y * BM;
  const int colBase = blockIdx.x * BN;

  if constexpr (MODE == MODE_SCORES) {
    // Tile entirely above the causal diagonal -> all -inf, skip GEMM.
    if (colBase > rowBase + BM - 1) {
      float* C = (float*)Cout;
      for (int i = tid; i < BM * BN; i += 256) {
        int r = i / BN, c = i - r * BN;
        C[(size_t)(rowBase + r) * ldc + colBase + c] = -__builtin_inff();
      }
      return;
    }
  }

  int kChunks = K / BK;
  if constexpr (MODE == MODE_CTX) {
    // keys beyond rowBase+BM-1 have zero probability: skip them
    int kmax = rowBase + BM;
    if (kmax > K) kmax = K;
    kChunks = (kmax + BK - 1) / BK;
  }

  // Per-thread staging coordinates: 2 threads per row, 16 bf16 (2x float4).
  const int ldr  = tid >> 1;            // 0..127
  const int ldcc = (tid & 1) * 16;      // 0 or 16
  const bf16_t* Ap = A  + (size_t)(rowBase + ldr) * lda + ldcc;
  const bf16_t* Bp = Bt + (size_t)(colBase + ldr) * ldb + ldcc;
  float4* const dA0 = reinterpret_cast<float4*>(&As[0][ldr * LDS_STRIDE + ldcc]);
  float4* const dB0 = reinterpret_cast<float4*>(&Bs[0][ldr * LDS_STRIDE + ldcc]);
  float4* const dA1 = reinterpret_cast<float4*>(&As[1][ldr * LDS_STRIDE + ldcc]);
  float4* const dB1 = reinterpret_cast<float4*>(&Bs[1][ldr * LDS_STRIDE + ldcc]);

  // Prologue: stage chunk 0 into buffer 0.
  {
    float4 a0 = ((const float4*)Ap)[0];
    float4 a1 = ((const float4*)Ap)[1];
    float4 b0 = ((const float4*)Bp)[0];
    float4 b1 = ((const float4*)Bp)[1];
    dA0[0] = a0; dA0[1] = a1;
    dB0[0] = b0; dB0[1] = b1;
  }
  __syncthreads();

  v8f zero = {0.f, 0.f, 0.f, 0.f, 0.f, 0.f, 0.f, 0.f};
  v8f acc[2][4];
  #pragma unroll
  for (int mi = 0; mi < 2; ++mi)
    #pragma unroll
    for (int ni = 0; ni < 4; ++ni) acc[mi][ni] = zero;

  union Frag { float4 f4[2]; v16bf v; };

  for (int kc = 0; kc < kChunks; ++kc) {
    const int buf = kc & 1;
    const bool haveNext = (kc + 1 < kChunks);

    // Issue next chunk's global loads early (overlap with WMMA below).
    float4 na0 = {}, na1 = {}, nb0 = {}, nb1 = {};
    if (haveNext) {
      const bf16_t* An = Ap + (size_t)(kc + 1) * BK;
      const bf16_t* Bn = Bp + (size_t)(kc + 1) * BK;
      na0 = ((const float4*)An)[0];
      na1 = ((const float4*)An)[1];
      nb0 = ((const float4*)Bn)[0];
      nb1 = ((const float4*)Bn)[1];
    }

    // A fragment (16x32 bf16): lane holds row = l16,
    //   lanes 0-15: K {0..7, 16..23}; lanes 16-31: K {8..15, 24..31}
    Frag a[2];
    #pragma unroll
    for (int mi = 0; mi < 2; ++mi) {
      const int row = waveM * 32 + mi * 16 + l16;
      const float4* p = reinterpret_cast<const float4*>(&As[buf][row * LDS_STRIDE]);
      a[mi].f4[0] = p[halfSel];       // halves 8*halfSel .. +8
      a[mi].f4[1] = p[2 + halfSel];   // halves 16+8*halfSel .. +8
    }
    // B fragment (32x16 bf16): lane holds col n = l16,
//   lanes 0-15: K {0..7, 16..23}; lanes 16-31: K {8..15, 24..31}
    Frag b[4];
    #pragma unroll
    for (int ni = 0; ni < 4; ++ni) {
      const int col = waveN * 64 + ni * 16 + l16;
      const float4* p = reinterpret_cast<const float4*>(&Bs[buf][col * LDS_STRIDE]);
      b[ni].f4[0] = p[halfSel];
      b[ni].f4[1] = p[2 + halfSel];
    }

    #pragma unroll
    for (int mi = 0; mi < 2; ++mi)
      #pragma unroll
      for (int ni = 0; ni < 4; ++ni)
        acc[mi][ni] = __builtin_amdgcn_wmma_f32_16x16x32_bf16(
            false, a[mi].v, false, b[ni].v, (short)0, acc[mi][ni], false, false);

    // Stage next chunk into the other buffer (readers of it synced last iter).
    if (haveNext) {
      float4* dA = buf ? dA0 : dA1;
      float4* dB = buf ? dB0 : dB1;
      dA[0] = na0; dA[1] = na1;
      dB[0] = nb0; dB[1] = nb1;
      __syncthreads();
    }
  }

  // Epilogue. C/D layout: VGPR r, lanes 0-15 -> row r, lanes 16-31 -> row r+8;
  // col = lane & 15.
  #pragma unroll
  for (int mi = 0; mi < 2; ++mi) {
    #pragma unroll
    for (int ni = 0; ni < 4; ++ni) {
      #pragma unroll
      for (int r = 0; r < 8; ++r) {
        const int m = rowBase + waveM * 32 + mi * 16 + r + halfSel * 8;
        const int n = colBase + waveN * 64 + ni * 16 + l16;
        const float v = acc[mi][ni][r];
        const size_t idx = (size_t)m * ldc + n;
        if constexpr (MODE == MODE_BF16 || MODE == MODE_CTX) {
          ((bf16_t*)Cout)[idx] = (bf16_t)v;
        } else if constexpr (MODE == MODE_BF16_TRANS) {
          ((bf16_t*)Cout)[(size_t)n * ldc + m] = (bf16_t)v;
        } else if constexpr (MODE == MODE_SCORES) {
          ((float*)Cout)[idx] = (n <= m) ? v * scale : -__builtin_inff();
        } else if constexpr (MODE == MODE_RESID) {
          ((float*)Cout)[idx] = v + resid[idx];
        } else if constexpr (MODE == MODE_RELU_BIAS) {
          float t = v + bias[n];
          ((bf16_t*)Cout)[idx] = (bf16_t)(t > 0.f ? t : 0.f);
        } else if constexpr (MODE == MODE_BIAS_RESID) {
          ((float*)Cout)[idx] = v + bias[n] + resid[idx];
        }
      }
    }
  }
}

// ---------------------------------------------------------------------------
// LayerNorm over D=1024, fp32 stats, bf16 output. One block (256 thr) per row.
// ---------------------------------------------------------------------------
__global__ __launch_bounds__(256)
void layernorm_bf16(const float* __restrict__ x,
                    const float* __restrict__ g,
                    const float* __restrict__ bta,
                    bf16_t* __restrict__ out)
{
  const int row = blockIdx.x;
  const float* xr = x + (size_t)row * DMODEL;

  float vals[4];
  float s = 0.f, ss = 0.f;
  #pragma unroll
  for (int j = 0; j < 4; ++j) {
    float v = xr[threadIdx.x + j * 256];
    vals[j] = v;
    s += v;
    ss += v * v;
  }
  #pragma unroll
  for (int off = 16; off > 0; off >>= 1) {
    s  += __shfl_xor(s, off, 32);
    ss += __shfl_xor(ss, off, 32);
  }
  __shared__ float sh[16];
  const int w = threadIdx.x >> 5, ln = threadIdx.x & 31;
  if (ln == 0) { sh[w] = s; sh[8 + w] = ss; }
  __syncthreads();
  s = 0.f; ss = 0.f;
  #pragma unroll
  for (int i = 0; i < 8; ++i) { s += sh[i]; ss += sh[8 + i]; }

  const float mu = s * (1.f / DMODEL);
  const float var = ss * (1.f / DMODEL) - mu * mu;
  const float rs = rsqrtf(var + 1e-5f);

  #pragma unroll
  for (int j = 0; j < 4; ++j) {
    const int c = threadIdx.x + j * 256;
    out[(size_t)row * DMODEL + c] = (bf16_t)((vals[j] - mu) * rs * g[c] + bta[c]);
  }
}

// ---------------------------------------------------------------------------
// Row softmax over SEQ=2048 fp32 scores -> bf16 probabilities (0 where -inf).
// One block (256 thr) per (b, q) row.
// ---------------------------------------------------------------------------
__global__ __launch_bounds__(256)
void softmax_row(const float* __restrict__ S, bf16_t* __restrict__ P)
{
  const int row = blockIdx.x;
  const float* sr = S + (size_t)row * SEQ;
  bf16_t* pr = P + (size_t)row * SEQ;

  float vals[8];
  float mx = -__builtin_inff();
  #pragma unroll
  for (int j = 0; j < 8; ++j) {
    float v = sr[threadIdx.x + j * 256];
    vals[j] = v;
    mx = fmaxf(mx, v);
  }
  #pragma unroll
  for (int off = 16; off > 0; off >>= 1) mx = fmaxf(mx, __shfl_xor(mx, off, 32));
  __shared__ float sh[8];
  const int w = threadIdx.x >> 5, ln = threadIdx.x & 31;
  if (ln == 0) sh[w] = mx;
  __syncthreads();
  mx = sh[0];
  #pragma unroll
  for (int i = 1; i < 8; ++i) mx = fmaxf(mx, sh[i]);
  __syncthreads();

  float sum = 0.f;
  #pragma unroll
  for (int j = 0; j < 8; ++j) {
    float e = __expf(vals[j] - mx);  // exp(-inf - mx) == 0
    vals[j] = e;
    sum += e;
  }
  #pragma unroll
  for (int off = 16; off > 0; off >>= 1) sum += __shfl_xor(sum, off, 32);
  if (ln == 0) sh[w] = sum;
  __syncthreads();
  sum = 0.f;
  #pragma unroll
  for (int i = 0; i < 8; ++i) sum += sh[i];
  const float inv = 1.f / sum;

  #pragma unroll
  for (int j = 0; j < 8; ++j)
    pr[threadIdx.x + j * 256] = (bf16_t)(vals[j] * inv);
}

// ---------------------------------------------------------------------------
// fp32 -> bf16 weight conversion
// ---------------------------------------------------------------------------
__global__ __launch_bounds__(256)
void cvt_f32_bf16(const float* __restrict__ in, bf16_t* __restrict__ out, int n)
{
  const int i = blockIdx.x * 256 + threadIdx.x;
  if (i < n) out[i] = (bf16_t)in[i];
}

// ---------------------------------------------------------------------------
// Host-side orchestration
// ---------------------------------------------------------------------------
extern "C" void kernel_launch(void* const* d_in, const int* in_sizes, int n_in,
                              void* d_out, int out_size, void* d_ws, size_t ws_size,
                              hipStream_t stream)
{
  (void)in_sizes; (void)n_in; (void)out_size; (void)ws_size;

  const float* x     = (const float*)d_in[0];
  const float* W_q   = (const float*)d_in[1];
  const float* W_k   = (const float*)d_in[2];
  const float* W_v   = (const float*)d_in[3];
  const float* W_o   = (const float*)d_in[4];
  const float* fc1_w = (const float*)d_in[5];
  const float* fc1_b = (const float*)d_in[6];
  const float* fc2_w = (const float*)d_in[7];
  const float* fc2_b = (const float*)d_in[8];
  const float* ln1_g = (const float*)d_in[9];
  const float* ln1_b = (const float*)d_in[10];
  const float* ln2_g = (const float*)d_in[11];
  const float* ln2_b = (const float*)d_in[12];
  float* out = (float*)d_out;

  // workspace carve-up (256B aligned)
  char* ws = (char*)d_ws;
  size_t off = 0;
  auto take = [&](size_t bytes) -> char* {
    char* p = ws + off;
    off = (off + bytes + 255) & ~(size_t)255;
    return p;
  };
  bf16_t* wq   = (bf16_t*)take((size_t)DMODEL * DMODEL * 2);
  bf16_t* wk   = (bf16_t*)take((size_t)DMODEL * DMODEL * 2);
  bf16_t* wv   = (bf16_t*)take((size_t)DMODEL * DMODEL * 2);
  bf16_t* wo   = (bf16_t*)take((size_t)DMODEL * DMODEL * 2);
  bf16_t* f1w  = (bf16_t*)take((size_t)DFF * DMODEL * 2);
  bf16_t* f2w  = (bf16_t*)take((size_t)DMODEL * DFF * 2);
  bf16_t* xn   = (bf16_t*)take((size_t)NTOK * DMODEL * 2);
  bf16_t* qb   = (bf16_t*)take((size_t)NTOK * DMODEL * 2);
  bf16_t* kb   = (bf16_t*)take((size_t)NTOK * DMODEL * 2);
  bf16_t* vT   = (bf16_t*)take((size_t)BATCH * DMODEL * SEQ * 2);
  float*  sc   = (float*) take((size_t)BATCH * SEQ * SEQ * 4);
  bf16_t* P    = (bf16_t*)take((size_t)BATCH * SEQ * SEQ * 2);
  bf16_t* ctx  = (bf16_t*)take((size_t)NTOK * DMODEL * 2);
  float*  x1   = (float*) take((size_t)NTOK * DMODEL * 4);
  bf16_t* xn2  = (bf16_t*)take((size_t)NTOK * DMODEL * 2);
  bf16_t* h    = (bf16_t*)take((size_t)NTOK * DFF * 2);

  const int nDD = DMODEL * DMODEL;
  const int nFD = DFF * DMODEL;
  auto cvtGrid = [](int n) { return dim3((n + 255) / 256); };
  cvt_f32_bf16<<<cvtGrid(nDD), 256, 0, stream>>>(W_q,   wq,  nDD);
  cvt_f32_bf16<<<cvtGrid(nDD), 256, 0, stream>>>(W_k,   wk,  nDD);
  cvt_f32_bf16<<<cvtGrid(nDD), 256, 0, stream>>>(W_v,   wv,  nDD);
  cvt_f32_bf16<<<cvtGrid(nDD), 256, 0, stream>>>(W_o,   wo,  nDD);
  cvt_f32_bf16<<<cvtGrid(nFD), 256, 0, stream>>>(fc1_w, f1w, nFD);
  cvt_f32_bf16<<<cvtGrid(nFD), 256, 0, stream>>>(fc2_w, f2w, nFD);

  // LN1
  layernorm_bf16<<<NTOK, 256, 0, stream>>>(x, ln1_g, ln1_b, xn);

  // Q, K projections: [NTOK x D] = xn @ W^T
  {
    dim3 grid(DMODEL / BN, NTOK / BM);
    gemm_wmma<MODE_BF16><<<grid, 256, 0, stream>>>(
        xn, DMODEL, wq, DMODEL, qb, DMODEL, nullptr, nullptr,
        NTOK, DMODEL, DMODEL, 1.f);
    gemm_wmma<MODE_BF16><<<grid, 256, 0, stream>>>(
        xn, DMODEL, wk, DMODEL, kb, DMODEL, nullptr, nullptr,
        NTOK, DMODEL, DMODEL, 1.f);
  }
  // V, stored transposed per batch: vT[b][d][s]
  for (int b = 0; b < BATCH; ++b) {
    dim3 grid(DMODEL / BN, SEQ / BM);
    gemm_wmma<MODE_BF16_TRANS><<<grid, 256, 0, stream>>>(
        xn + (size_t)b * SEQ * DMODEL, DMODEL, wv, DMODEL,
        vT + (size_t)b * DMODEL * SEQ, SEQ, nullptr, nullptr,
        SEQ, DMODEL, DMODEL, 1.f);
  }
  // causal scores per batch: sc[b][q][k] = (q.k)/sqrt(D), -inf above diag
  for (int b = 0; b < BATCH; ++b) {
    dim3 grid(SEQ / BN, SEQ / BM);
    gemm_wmma<MODE_SCORES><<<grid, 256, 0, stream>>>(
        qb + (size_t)b * SEQ * DMODEL, DMODEL,
        kb + (size_t)b * SEQ * DMODEL, DMODEL,
        sc + (size_t)b * SEQ * SEQ, SEQ, nullptr, nullptr,
        SEQ, SEQ, DMODEL, 0.03125f /* 1/sqrt(1024) */);
  }
  // softmax rows -> bf16 probabilities
  softmax_row<<<BATCH * SEQ, 256, 0, stream>>>(sc, P);

  // ctx[b] = P[b] @ V[b]  (Bt = vT rows, causal K-limit per row block)
  for (int b = 0; b < BATCH; ++b) {
    dim3 grid(DMODEL / BN, SEQ / BM);
    gemm_wmma<MODE_CTX><<<grid, 256, 0, stream>>>(
        P + (size_t)b * SEQ * SEQ, SEQ,
        vT + (size_t)b * DMODEL * SEQ, SEQ,
        ctx + (size_t)b * SEQ * DMODEL, DMODEL, nullptr, nullptr,
        SEQ, DMODEL, SEQ, 1.f);
  }
  // attn_out = ctx @ W_o^T ; x1 = x + attn_out (fp32)
  {
    dim3 grid(DMODEL / BN, NTOK / BM);
    gemm_wmma<MODE_RESID><<<grid, 256, 0, stream>>>(
        ctx, DMODEL, wo, DMODEL, x1, DMODEL, nullptr, x,
        NTOK, DMODEL, DMODEL, 1.f);
  }
  // LN2
  layernorm_bf16<<<NTOK, 256, 0, stream>>>(x1, ln2_g, ln2_b, xn2);

  // h = relu(xn2 @ fc1_w^T + fc1_b)
  {
    dim3 grid(DFF / BN, NTOK / BM);
    gemm_wmma<MODE_RELU_BIAS><<<grid, 256, 0, stream>>>(
        xn2, DMODEL, f1w, DMODEL, h, DFF, fc1_b, nullptr,
        NTOK, DFF, DMODEL, 1.f);
  }
  // out = x1 + h @ fc2_w^T + fc2_b
  {
    dim3 grid(DMODEL / BN, NTOK / BM);
    gemm_wmma<MODE_BIAS_RESID><<<grid, 256, 0, stream>>>(
        h, DFF, f2w, DFF, out, DMODEL, fc2_b, x1,
        NTOK, DMODEL, DFF, 1.f);
  }
}
